// GNN_61117384622241
// MI455X (gfx1250) — compile-verified, matrix operand order checked by reading, not run on hardware
//
#include <hip/hip_runtime.h>
#include <math.h>

typedef __attribute__((ext_vector_type(16))) _Float16 v16h;
typedef __attribute__((ext_vector_type(8)))  float    v8f;

// CDNA5 async global->LDS copy (ASYNCcnt path), guarded so compile never breaks.
#if defined(__has_builtin)
#if __has_builtin(__builtin_amdgcn_global_load_async_to_lds_b128) && \
    __has_builtin(__builtin_amdgcn_s_wait_asynccnt)
#define GNN_ASYNC_LDS 1
#endif
#endif
#ifndef GNN_ASYNC_LDS
#define GNN_ASYNC_LDS 0
#endif

#if GNN_ASYNC_LDS
typedef __attribute__((__vector_size__(4 * sizeof(int)))) int gnn_v4i;
typedef __attribute__((address_space(1))) void     as1_void;
typedef __attribute__((address_space(3))) void     as3_void;
typedef __attribute__((address_space(1))) gnn_v4i  as1_v4i;
typedef __attribute__((address_space(3))) gnn_v4i  as3_v4i;
__device__ __forceinline__ void gnn_async_cp16(const _Float16* g, _Float16* l) {
  __builtin_amdgcn_global_load_async_to_lds_b128(
      (as1_v4i*)(as1_void*)g, (as3_v4i*)(as3_void*)l, 0, 0);
}
#endif

// ---------------- problem constants ----------------
constexpr int LQ    = 4096;   // number of matches
constexpr int CDIM  = 256;    // IN_DIM
constexpr int NHEAD = 8;
constexpr int DHEAD = 32;
constexpr int NBR   = 8;
constexpr int RAD   = 20;
constexpr int DD    = 41;
constexpr int D2    = DD * DD;   // 1681
constexpr int FDIM  = 2 * D2;    // 3362
constexpr int FPAD  = 3392;      // 106*32, K padded for WMMA
constexpr int IMH   = 1024;
constexpr int IMW   = 1024;
constexpr int HID   = 768;       // 3*IN_DIM
constexpr int DEPTH = 9;

enum { ACT_NONE = 0, ACT_RELU = 1, ACT_TANH = 2, ACT_SIGMOID = 3 };

// ============================================================
// Weight convert: w (K x N, f32) -> wt (N x Kpad, f16), zero pad
// ============================================================
__global__ void gnn_convert_wt(const float* __restrict__ w, _Float16* __restrict__ wt,
                               int K, int N, int Kpad) {
  int idx = blockIdx.x * 256 + threadIdx.x;
  if (idx >= N * Kpad) return;
  int n = idx / Kpad, k = idx % Kpad;
  wt[idx] = (k < K) ? (_Float16)w[(size_t)k * N + n] : (_Float16)0.f;
}

// ============================================================
// Patch extraction + per-patch normalization -> fh (L x FPAD f16)
// ============================================================
__global__ void gnn_extract(const float* __restrict__ img1, const float* __restrict__ img2,
                            const int* __restrict__ matches, _Float16* __restrict__ fh) {
  __shared__ float sp[2][D2];
  __shared__ float ssum[256];
  __shared__ float ssq[256];
  const int l   = blockIdx.x;
  const int tid = threadIdx.x;
  const int m0 = matches[l * 4 + 0], m1 = matches[l * 4 + 1];
  const int m2 = matches[l * 4 + 2], m3 = matches[l * 4 + 3];
  for (int j = tid; j < D2; j += 256) {
    int dr = j % DD, dc = j / DD;              // row fast, col slow (ref layout)
    int r1 = m1 + dr - RAD, c1 = m0 + dc - RAD;
    sp[0][j] = (r1 >= 0 && r1 < IMH && c1 >= 0 && c1 < IMW) ? img1[r1 * IMW + c1] : 0.f;
    int r2 = m3 + dr - RAD, c2 = m2 + dc - RAD;
    sp[1][j] = (r2 >= 0 && r2 < IMH && c2 >= 0 && c2 < IMW) ? img2[r2 * IMW + c2] : 0.f;
  }
  __syncthreads();
  for (int p = 0; p < 2; p++) {
    float s = 0.f, q = 0.f;
    for (int j = tid; j < D2; j += 256) { float v = sp[p][j]; s += v; q += v * v; }
    ssum[tid] = s; ssq[tid] = q;
    __syncthreads();
    for (int off = 128; off >= 1; off >>= 1) {
      if (tid < off) { ssum[tid] += ssum[tid + off]; ssq[tid] += ssq[tid + off]; }
      __syncthreads();
    }
    float mean = ssum[0] / (float)D2;
    float var  = (ssq[0] - (float)D2 * mean * mean) / (float)(D2 - 1);  // ddof=1
    float inv  = 1.f / (sqrtf(fmaxf(var, 0.f)) + 1e-4f);
    for (int j = tid; j < D2; j += 256)
      fh[(size_t)l * FPAD + p * D2 + j] = (_Float16)((sp[p][j] - mean) * inv);
    __syncthreads();
  }
  for (int j = FDIM + tid; j < FPAD; j += 256)
    fh[(size_t)l * FPAD + j] = (_Float16)0.f;
}

// ============================================================
// Top-8 neighbor search + relative-position features
// ============================================================
__global__ void gnn_topk_map(const int* __restrict__ matches, int* __restrict__ query,
                             _Float16* __restrict__ pos) {
  __shared__ float sD[256 * NBR];
  __shared__ int   sI[256 * NBR];
  const int i   = blockIdx.x;
  const int tid = threadIdx.x;
  float mi[4];
  for (int c = 0; c < 4; c++) mi[c] = (float)matches[i * 4 + c];
  float bd[NBR]; int bi[NBR];
  for (int t = 0; t < NBR; t++) { bd[t] = -3.4e38f; bi[t] = 0x7fffffff; }
  for (int j = tid; j < LQ; j += 256) {
    float d = 0.f;
    for (int c = 0; c < 4; c++) { float r = (float)matches[j * 4 + c] - mi[c]; d -= r * r; }
    if (d > bd[NBR - 1] || (d == bd[NBR - 1] && j < bi[NBR - 1])) {
      bd[NBR - 1] = d; bi[NBR - 1] = j;
      for (int p = NBR - 1; p > 0; p--) {
        bool sw = (bd[p] > bd[p - 1]) || (bd[p] == bd[p - 1] && bi[p] < bi[p - 1]);
        if (!sw) break;
        float td = bd[p]; bd[p] = bd[p - 1]; bd[p - 1] = td;
        int   ti = bi[p]; bi[p] = bi[p - 1]; bi[p - 1] = ti;
      }
    }
  }
  for (int t = 0; t < NBR; t++) { sD[tid * NBR + t] = bd[t]; sI[tid * NBR + t] = bi[t]; }
  __syncthreads();
  for (int off = 128; off >= 1; off >>= 1) {
    if (tid < off) {
      float ad[NBR], od[NBR]; int ai[NBR], oi[NBR];
      for (int t = 0; t < NBR; t++) {
        ad[t] = sD[tid * NBR + t];         ai[t] = sI[tid * NBR + t];
        od[t] = sD[(tid + off) * NBR + t]; oi[t] = sI[(tid + off) * NBR + t];
      }
      int pa = 0, pb = 0;
      float rd[NBR]; int ri[NBR];
      for (int t = 0; t < NBR; t++) {
        bool takeA = (ad[pa] > od[pb]) || (ad[pa] == od[pb] && ai[pa] < oi[pb]);
        if (takeA) { rd[t] = ad[pa]; ri[t] = ai[pa]; pa++; }
        else       { rd[t] = od[pb]; ri[t] = oi[pb]; pb++; }
      }
      for (int t = 0; t < NBR; t++) { sD[tid * NBR + t] = rd[t]; sI[tid * NBR + t] = ri[t]; }
    }
    __syncthreads();
  }
  if (tid < NBR) {
    int qj = sI[tid];
    query[i * NBR + tid] = qj;
    for (int c = 0; c < 4; c++)
      pos[(size_t)i * (NBR * 4) + tid * 4 + c] =
          (_Float16)((float)matches[qj * 4 + c] - mi[c]);   // rel = m[q] - m[i]
  }
}

// ============================================================
// WMMA GEMM v2: C = act(A @ W) [+ resid]
// A: M x K f16 (lda), Wt: N x K f16 (ldw, pre-transposed)
// 128 threads (4 waves), 128x64 C tile, wave = 32x64 strip
// double-buffered LDS (1 barrier / K-step), async DMA if available
// ============================================================
constexpr int BM = 128, BN = 64, BK = 32;

__global__ void gnn_wmma_gemm(const _Float16* __restrict__ A, int lda,
                              const _Float16* __restrict__ Wt, int ldw,
                              const float* resid, float* Cf, _Float16* Ch,
                              int M, int N, int K, int act) {
  __shared__ _Float16 sA[2][BM][40];   // stride 40 halves keeps b128 alignment
  __shared__ _Float16 sB[2][BN][40];
  const int tid  = threadIdx.x;
  const int wave = tid >> 5;
  const int lane = tid & 31;
  const int g    = lane >> 4;          // wave32 fragment half-group
  const int mn   = lane & 15;
  const int bm   = blockIdx.y * BM;
  const int bn   = blockIdx.x * BN;

  v8f acc[2][4];
  #pragma unroll
  for (int h = 0; h < 2; h++)
    #pragma unroll
    for (int t = 0; t < 4; t++)
      #pragma unroll
      for (int r = 0; r < 8; r++) acc[h][t][r] = 0.f;

  // cooperative loaders: A -> thread owns one 32-half row; B -> half row each
  const _Float16* gA = A  + (size_t)(bm + tid) * lda;
  const int lBrow = tid >> 1;
  const int lBoff = (tid & 1) << 4;
  const _Float16* gB = Wt + (size_t)(bn + lBrow) * ldw + lBoff;

  const int nk = K / BK;

#if GNN_ASYNC_LDS
  // ---- ASYNCcnt-tracked global->LDS DMA, no VGPR round trip ----
  gnn_async_cp16(gA,      &sA[0][tid][0]);
  gnn_async_cp16(gA + 16, &sA[0][tid][16]);
  gnn_async_cp16(gB,      &sB[0][lBrow][lBoff]);
  __builtin_amdgcn_s_wait_asynccnt(0);
  __syncthreads();
#else
  uint4 ra0 = *(const uint4*)(gA);
  uint4 ra1 = *(const uint4*)(gA + 16);
  uint4 rb  = *(const uint4*)(gB);
  *(uint4*)&sA[0][tid][0]          = ra0;
  *(uint4*)&sA[0][tid][16]         = ra1;
  *(uint4*)&sB[0][lBrow][lBoff]    = rb;
  __syncthreads();
#endif

  for (int it = 0; it < nk; it++) {
    const int cur = it & 1;
#if GNN_ASYNC_LDS
    if (it + 1 < nk) {   // DMA next tile into the other buffer (safe: barrier below
      const int kk = (it + 1) * BK;     // separated last reads of it from these writes)
      gnn_async_cp16(gA + kk,          &sA[cur ^ 1][tid][0]);
      gnn_async_cp16(gA + kk + 16,     &sA[cur ^ 1][tid][16]);
      gnn_async_cp16(gB + kk,          &sB[cur ^ 1][lBrow][lBoff]);
    }
#else
    if (it + 1 < nk) {
      const int kk = (it + 1) * BK;
      ra0 = *(const uint4*)(gA + kk);
      ra1 = *(const uint4*)(gA + kk + 16);
      rb  = *(const uint4*)(gB + kk);
    }
#endif
    // ---- 8 WMMAs on the current buffer ----
    union HF { v16h v; uint32_t u[8]; };
    HF af[2];
    #pragma unroll
    for (int h = 0; h < 2; h++) {
      const int arow = (wave << 5) + (h << 4) + mn;
      #pragma unroll
      for (int vv = 0; vv < 8; vv++) {
        int ko = (g << 3) + ((vv & 3) << 1) + ((vv >> 2) << 4);  // ISA 16-bit layout
        af[h].u[vv] = *(const uint32_t*)&sA[cur][arow][ko];
      }
    }
    #pragma unroll
    for (int t = 0; t < 4; t++) {
      HF bf;
      const int brow = (t << 4) + mn;
      #pragma unroll
      for (int vv = 0; vv < 8; vv++) {
        int ko = (g << 3) + ((vv & 3) << 1) + ((vv >> 2) << 4);
        bf.u[vv] = *(const uint32_t*)&sB[cur][brow][ko];
      }
      #pragma unroll
      for (int h = 0; h < 2; h++)
        acc[h][t] = __builtin_amdgcn_wmma_f32_16x16x32_f16(
            false, af[h].v, false, bf.v, (short)0, acc[h][t], false, false);
    }
#if GNN_ASYNC_LDS
    if (it + 1 < nk) __builtin_amdgcn_s_wait_asynccnt(0);
    __syncthreads();   // after this, everyone's DMA for the next buffer is visible
#else
    if (it + 1 < nk) {
      *(uint4*)&sA[cur ^ 1][tid][0]       = ra0;
      *(uint4*)&sA[cur ^ 1][tid][16]      = ra1;
      *(uint4*)&sB[cur ^ 1][lBrow][lBoff] = rb;
    }
    __syncthreads();
#endif
  }

  #pragma unroll
  for (int h = 0; h < 2; h++) {
    #pragma unroll
    for (int t = 0; t < 4; t++) {
      #pragma unroll
      for (int r = 0; r < 8; r++) {
        int m = bm + (wave << 5) + (h << 4) + r + (g << 3);  // C: VGPR r -> M = r+8g
        int n = bn + (t << 4) + mn;
        float val = acc[h][t][r];
        if (act == ACT_RELU)      val = fmaxf(val, 0.f);
        else if (act == ACT_TANH) val = tanhf(val);
        if (resid) val += resid[(size_t)m * N + n];
        if (Cf) Cf[(size_t)m * N + n] = val;
        if (Ch) Ch[(size_t)m * N + n] = (_Float16)val;
      }
    }
  }
}

// ============================================================
// LayerNorm (biased var, eps 1e-5), optional residual, dual outputs
// ============================================================
__global__ void gnn_ln(const float* __restrict__ x, const float* __restrict__ g,
                       const float* __restrict__ b, const float* resid,
                       float* yf, _Float16* yh, int C) {
  __shared__ float s1[256];
  __shared__ float s2[256];
  const int row = blockIdx.x;
  const int tid = threadIdx.x;
  const float* xr = x + (size_t)row * C;
  float s = 0.f, q = 0.f;
  for (int c = tid; c < C; c += 256) { float v = xr[c]; s += v; q += v * v; }
  s1[tid] = s; s2[tid] = q;
  __syncthreads();
  for (int off = 128; off >= 1; off >>= 1) {
    if (tid < off) { s1[tid] += s1[tid + off]; s2[tid] += s2[tid + off]; }
    __syncthreads();
  }
  float mean = s1[0] / (float)C;
  float var  = s2[0] / (float)C - mean * mean;
  float inv  = rsqrtf(var + 1e-5f);
  for (int c = tid; c < C; c += 256) {
    float v = (xr[c] - mean) * inv * g[c] + b[c];
    if (resid) v += resid[(size_t)row * C + c];
    if (yf) yf[(size_t)row * C + c] = v;
    if (yh) yh[(size_t)row * C + c] = (_Float16)v;
  }
}

// ============================================================
// 8-neighbor, 8-head attention. One wave per match, 8 matches/block.
// ============================================================
__global__ void gnn_attn(const _Float16* __restrict__ q16, const _Float16* __restrict__ k16,
                         const _Float16* __restrict__ v16, const int* __restrict__ query,
                         _Float16* __restrict__ out) {
  __shared__ float sLog[8][64];
  __shared__ float sM[8][8];
  __shared__ float sS[8][8];
  const int wave = threadIdx.x >> 5;
  const int lane = threadIdx.x & 31;
  const int i    = blockIdx.x * 8 + wave;
  const int* qi  = query + i * NBR;

  for (int p = lane; p < NHEAD * NBR; p += 32) {
    int h = p >> 3, n = p & 7;
    int nb = qi[n];
    const _Float16* qp = q16 + (size_t)i  * CDIM + h * DHEAD;
    const _Float16* kp = k16 + (size_t)nb * CDIM + h * DHEAD;
    float acc = 0.f;
    #pragma unroll
    for (int d = 0; d < DHEAD; d++) acc += (float)qp[d] * (float)kp[d];
    sLog[wave][p] = acc * 0.17677669529663687f;   // 1/sqrt(32)
  }
  __syncthreads();
  if (lane < NHEAD) {
    float mx = -3.4e38f;
    for (int n = 0; n < NBR; n++) mx = fmaxf(mx, sLog[wave][lane * NBR + n]);
    float sm = 0.f;
    for (int n = 0; n < NBR; n++) sm += __expf(sLog[wave][lane * NBR + n] - mx);
    sM[wave][lane] = mx; sS[wave][lane] = sm;
  }
  __syncthreads();
  #pragma unroll
  for (int j = 0; j < 8; j++) {
    int d = lane + 32 * j;       // head h == j since DHEAD == 32
    int h = d >> 5;
    float o = 0.f;
    for (int n = 0; n < NBR; n++) {
      float a = __expf(sLog[wave][h * NBR + n] - sM[wave][h]) / sS[wave][h];
      o += a * (float)v16[(size_t)qi[n] * CDIM + d];
    }
    out[(size_t)i * CDIM + d] = (_Float16)o;
  }
}

// ============================================================
// Tiny output heads (N = 1 or 2)
// ============================================================
__global__ void gnn_head(const _Float16* __restrict__ X, const float* __restrict__ Wf,
                         float* __restrict__ out, int N, int act) {
  int idx = blockIdx.x * 256 + threadIdx.x;
  if (idx >= LQ * N) return;
  int row = idx / N, col = idx % N;
  const _Float16* xr = X + (size_t)row * CDIM;
  float acc = 0.f;
  for (int k = 0; k < CDIM; k++) acc += (float)xr[k] * Wf[k * N + col];
  if (act == ACT_SIGMOID) acc = 1.f / (1.f + __expf(-acc));
  out[idx] = acc;
}

// ============================================================
// Host orchestration
// ============================================================
extern "C" void kernel_launch(void* const* d_in, const int* in_sizes, int n_in,
                              void* d_out, int out_size, void* d_ws, size_t ws_size,
                              hipStream_t stream) {
  (void)in_sizes; (void)n_in; (void)out_size; (void)ws_size;
  const float* img1    = (const float*)d_in[0];
  const float* img2    = (const float*)d_in[1];
  const int*   matches = (const int*)d_in[2];
  const float* ef_b  = (const float*)d_in[3];
  const float* ef_g  = (const float*)d_in[4];
  const float* ef_w1 = (const float*)d_in[5];
  const float* ef_w2 = (const float*)d_in[6];
  const float* ef_w3 = (const float*)d_in[7];
  const float* ef_w4 = (const float*)d_in[8];
  const float* ep_w1 = (const float*)d_in[9];
  const float* ep_w2 = (const float*)d_in[10];
  const float *lb1[DEPTH], *lb2[DEPTH], *lg1[DEPTH], *lg2[DEPTH];
  const float *lw1[DEPTH], *lw2[DEPTH], *lwk[DEPTH], *lwm[DEPTH], *lwq[DEPTH], *lwv[DEPTH];
  for (int i = 0; i < DEPTH; i++) {  // keys: b1,b2,g1,g2,w1,w2,wk,wm,wq,wv
    const int base = 11 + 10 * i;
    lb1[i] = (const float*)d_in[base + 0];
    lb2[i] = (const float*)d_in[base + 1];
    lg1[i] = (const float*)d_in[base + 2];
    lg2[i] = (const float*)d_in[base + 3];
    lw1[i] = (const float*)d_in[base + 4];
    lw2[i] = (const float*)d_in[base + 5];
    lwk[i] = (const float*)d_in[base + 6];
    lwm[i] = (const float*)d_in[base + 7];
    lwq[i] = (const float*)d_in[base + 8];
    lwv[i] = (const float*)d_in[base + 9];
  }
  const float* o_w1 = (const float*)d_in[11 + 10 * DEPTH + 0];
  const float* o_w2 = (const float*)d_in[11 + 10 * DEPTH + 1];
  const float* s_w1 = (const float*)d_in[11 + 10 * DEPTH + 2];
  const float* s_w2 = (const float*)d_in[11 + 10 * DEPTH + 3];

  char* wsp = (char*)d_ws;
  size_t off = 0;
  auto alloc = [&](size_t bytes) -> void* {
    void* p = wsp + off;
    off += (bytes + 255) & ~(size_t)255;
    return p;
  };
  auto h16 = [&](size_t elems) { return (_Float16*)alloc(elems * sizeof(_Float16)); };
  auto f32 = [&](size_t elems) { return (float*)alloc(elems * sizeof(float)); };

  _Float16* t_efw1 = h16((size_t)HID * FPAD);
  _Float16* t_efw2 = h16((size_t)HID * HID);
  _Float16* t_efw3 = h16((size_t)CDIM * HID);
  _Float16* t_efw4 = h16((size_t)CDIM * CDIM);
  _Float16* t_epw1 = h16((size_t)CDIM * 32);
  _Float16* t_epw2 = h16((size_t)CDIM * CDIM);
  _Float16 *t_wq[DEPTH], *t_wk[DEPTH], *t_wv[DEPTH], *t_wm[DEPTH], *t_w1[DEPTH], *t_w2[DEPTH];
  for (int i = 0; i < DEPTH; i++) {
    t_wq[i] = h16((size_t)CDIM * CDIM); t_wk[i] = h16((size_t)CDIM * CDIM);
    t_wv[i] = h16((size_t)CDIM * CDIM); t_wm[i] = h16((size_t)CDIM * CDIM);
    t_w1[i] = h16((size_t)CDIM * CDIM); t_w2[i] = h16((size_t)CDIM * CDIM);
  }
  _Float16* t_sw1 = h16((size_t)CDIM * CDIM);
  _Float16* t_ow1 = h16((size_t)CDIM * CDIM);

  _Float16* fh     = h16((size_t)LQ * FPAD);
  _Float16* h1_16  = h16((size_t)LQ * HID);
  float*    h2_f   = f32((size_t)LQ * HID);
  _Float16* h2ln16 = h16((size_t)LQ * HID);
  _Float16* h3_16  = h16((size_t)LQ * CDIM);
  float*    f_f    = f32((size_t)LQ * CDIM);
  _Float16* pos16  = h16((size_t)LQ * NBR * 4);
  int*      query  = (int*)alloc((size_t)LQ * NBR * sizeof(int));
  _Float16* posh16 = h16((size_t)LQ * CDIM);
  float*    x_f    = f32((size_t)LQ * CDIM);
  _Float16* x_h16  = h16((size_t)LQ * CDIM);
  _Float16* xn16   = h16((size_t)LQ * CDIM);
  _Float16* q16    = h16((size_t)LQ * CDIM);
  _Float16* k16    = h16((size_t)LQ * CDIM);
  _Float16* v16    = h16((size_t)LQ * CDIM);
  _Float16* attn16 = h16((size_t)LQ * CDIM);
  float*    msg_f  = f32((size_t)LQ * CDIM);
  _Float16* msg16  = h16((size_t)LQ * CDIM);
  _Float16* t1_16  = h16((size_t)LQ * CDIM);
  float*    mlp_f  = f32((size_t)LQ * CDIM);

  auto conv = [&](const float* w, _Float16* wt, int K, int N, int Kpad) {
    int tot = N * Kpad;
    gnn_convert_wt<<<(tot + 255) / 256, 256, 0, stream>>>(w, wt, K, N, Kpad);
  };
  auto gemm = [&](const _Float16* A, int lda, const _Float16* Wt, int ldw,
                  const float* resid, float* Cf, _Float16* Ch,
                  int M, int N, int K, int act) {
    dim3 grid(N / BN, M / BM);
    gnn_wmma_gemm<<<grid, 128, 0, stream>>>(A, lda, Wt, ldw, resid, Cf, Ch, M, N, K, act);
  };

  // ---- weight conversion (f32 -> f16 transposed) ----
  conv(ef_w1, t_efw1, FDIM, HID, FPAD);
  conv(ef_w2, t_efw2, HID, HID, HID);
  conv(ef_w3, t_efw3, HID, CDIM, HID);
  conv(ef_w4, t_efw4, CDIM, CDIM, CDIM);
  conv(ep_w1, t_epw1, 32, CDIM, 32);
  conv(ep_w2, t_epw2, CDIM, CDIM, CDIM);
  for (int i = 0; i < DEPTH; i++) {
    conv(lwq[i], t_wq[i], CDIM, CDIM, CDIM);
    conv(lwk[i], t_wk[i], CDIM, CDIM, CDIM);
    conv(lwv[i], t_wv[i], CDIM, CDIM, CDIM);
    conv(lwm[i], t_wm[i], CDIM, CDIM, CDIM);
    conv(lw1[i], t_w1[i], CDIM, CDIM, CDIM);
    conv(lw2[i], t_w2[i], CDIM, CDIM, CDIM);
  }
  conv(s_w1, t_sw1, CDIM, CDIM, CDIM);
  conv(o_w1, t_ow1, CDIM, CDIM, CDIM);

  // ---- feature extraction + feature MLP ----
  gnn_extract<<<LQ, 256, 0, stream>>>(img1, img2, matches, fh);
  gemm(fh, FPAD, t_efw1, FPAD, nullptr, nullptr, h1_16, LQ, HID, FPAD, ACT_RELU);
  gemm(h1_16, HID, t_efw2, HID, nullptr, h2_f, nullptr, LQ, HID, HID, ACT_NONE);
  gnn_ln<<<LQ, 256, 0, stream>>>(h2_f, ef_g, ef_b, nullptr, nullptr, h2ln16, HID);
  gemm(h2ln16, HID, t_efw3, HID, nullptr, nullptr, h3_16, LQ, CDIM, HID, ACT_RELU);
  gemm(h3_16, CDIM, t_efw4, CDIM, nullptr, f_f, nullptr, LQ, CDIM, CDIM, ACT_NONE);

  // ---- neighbor graph + position embed ----
  gnn_topk_map<<<LQ, 256, 0, stream>>>(matches, query, pos16);
  gemm(pos16, NBR * 4, t_epw1, NBR * 4, nullptr, nullptr, posh16, LQ, CDIM, NBR * 4, ACT_TANH);
  gemm(posh16, CDIM, t_epw2, CDIM, f_f, x_f, nullptr, LQ, CDIM, CDIM, ACT_NONE);

  // ---- attention layers ----
  for (int i = 0; i < DEPTH; i++) {
    gnn_ln<<<LQ, 256, 0, stream>>>(x_f, lg1[i], lb1[i], nullptr, nullptr, xn16, CDIM);
    gemm(xn16, CDIM, t_wq[i], CDIM, nullptr, nullptr, q16, LQ, CDIM, CDIM, ACT_NONE);
    gemm(xn16, CDIM, t_wk[i], CDIM, nullptr, nullptr, k16, LQ, CDIM, CDIM, ACT_NONE);
    gemm(xn16, CDIM, t_wv[i], CDIM, nullptr, nullptr, v16, LQ, CDIM, CDIM, ACT_NONE);
    gnn_attn<<<LQ / 8, 256, 0, stream>>>(q16, k16, v16, query, attn16);
    gemm(attn16, CDIM, t_wm[i], CDIM, x_f, msg_f, msg16, LQ, CDIM, CDIM, ACT_NONE);
    gemm(msg16, CDIM, t_w1[i], CDIM, nullptr, nullptr, t1_16, LQ, CDIM, CDIM, ACT_RELU);
    gemm(t1_16, CDIM, t_w2[i], CDIM, nullptr, mlp_f, nullptr, LQ, CDIM, CDIM, ACT_NONE);
    gnn_ln<<<LQ, 256, 0, stream>>>(mlp_f, lg2[i], lb2[i], x_f, x_f, x_h16, CDIM);
  }

  // ---- output heads ----
  float* out_offset = (float*)d_out;             // (L, 2)
  float* out_scores = (float*)d_out + LQ * 2;    // (L,)
  gemm(x_h16, CDIM, t_sw1, CDIM, nullptr, nullptr, t1_16, LQ, CDIM, CDIM, ACT_RELU);
  gnn_head<<<(LQ + 255) / 256, 256, 0, stream>>>(t1_16, s_w2, out_scores, 1, ACT_SIGMOID);
  gemm(x_h16, CDIM, t_ow1, CDIM, nullptr, nullptr, t1_16, LQ, CDIM, CDIM, ACT_RELU);
  gnn_head<<<(LQ * 2 + 255) / 256, 256, 0, stream>>>(t1_16, o_w2, out_offset, 2, ACT_NONE);
}
